// Decoder_6725918786018
// MI455X (gfx1250) — compile-verified, matrix operand order checked by reading, not run on hardware
//
#include <hip/hip_runtime.h>

typedef _Float16 half_t;
typedef _Float16 v16h __attribute__((ext_vector_type(16)));
typedef float    v8f  __attribute__((ext_vector_type(8)));

#define V_    32000
#define VT_   32100
#define E_    256
#define H_    256
#define BS_   32
#define ML_   64     // max_len (decode steps)
#define SEQ_  128    // encoder length
#define LG_   32128  // V_ + SEQ_  (logits row)
#define TWOH_ 512
#define G3_   768    // 3*H
#define GIN_  768    // E + 2H

// Map (tile-row r in [0,16), k in [0,256)) -> packed half index inside one
// 16x256 operand slab laid out per the CDNA5 16-bit WMMA operand tables:
// 8 K-chunks of 32; within a chunk, lane = r + 16*g, element e with
// k%32 = (e<8 ? e : e+8) + 8*g.
__device__ __forceinline__ int pack_idx(int tile, int r, int k) {
    int kc  = k >> 5;
    int kk  = k & 31;
    int g   = (kk >> 3) & 1;
    int rem = kk - (g << 3);        // in {0..7} U {16..23}
    int e   = (rem < 8) ? rem : (rem - 8);
    int lane = r + (g << 4);
    return (((tile * 8 + kc) * 32) + lane) * 16 + e;
}

// ---- once-per-call kernels -------------------------------------------------

__global__ __launch_bounds__(256) void k_pack_wg(const float* __restrict__ Wg,
                                                 half_t* __restrict__ WgB) {
    int tid = blockIdx.x * 256 + threadIdx.x;       // < 32000*256
    int v = tid >> 8, k = tid & 255;
    WgB[pack_idx(v >> 4, v & 15, k)] = (half_t)Wg[tid];
}

__global__ __launch_bounds__(256) void k_enc_scores(const float* __restrict__ encoded,
                                                    const float* __restrict__ Wo_w,
                                                    const float* __restrict__ Wo_b,
                                                    float* __restrict__ es) {
    int bs = blockIdx.x;            // b*SEQ+s
    int j  = threadIdx.x;           // 0..255
    __shared__ float erow[TWOH_];
    const float* ep = encoded + (size_t)bs * TWOH_;
    erow[j] = ep[j]; erow[j + 256] = ep[j + 256];
    __syncthreads();
    const float* w = Wo_w + (size_t)j * TWOH_;
    float acc = Wo_b[j];
    for (int k = 0; k < TWOH_; ++k) acc += erow[k] * w[k];
    es[(size_t)bs * H_ + j] = tanhf(acc);
}

__global__ __launch_bounds__(256) void k_init(const float* __restrict__ encoded,
                                              const float* __restrict__ bridge_w,
                                              float* __restrict__ s0,
                                              float* __restrict__ ctx) {
    int tid = blockIdx.x * 256 + threadIdx.x;       // < 8192 + 16384
    if (tid < BS_ * H_) {
        int b = tid >> 8, j = tid & 255;
        const float* er = encoded + ((size_t)b * SEQ_ + (SEQ_ - 1)) * TWOH_;
        const float* w  = bridge_w + (size_t)j * TWOH_;
        float acc = 0.f;
        for (int k = 0; k < TWOH_; ++k) acc += er[k] * w[k];
        s0[tid] = acc;
    } else {
        ctx[tid - BS_ * H_] = 0.f;
    }
}

// ---- per-step kernels ------------------------------------------------------

__global__ __launch_bounds__(256) void k_gru(const int* __restrict__ inputs, int t,
                                             const float* __restrict__ embed,
                                             const float* __restrict__ ctx,
                                             const float* __restrict__ s_prev,
                                             const float* __restrict__ w_ih,
                                             const float* __restrict__ w_hh,
                                             const float* __restrict__ b_ih,
                                             const float* __restrict__ b_hh,
                                             float* __restrict__ gi,
                                             float* __restrict__ gh) {
    int tid = blockIdx.x * 256 + threadIdx.x;       // < 32*768
    int b = tid / G3_, j = tid % G3_;
    int tok = inputs[b * ML_ + t];
    const float* x  = embed + (size_t)tok * E_;
    const float* c  = ctx + (size_t)b * TWOH_;
    const float* wi = w_ih + (size_t)j * GIN_;
    float a = b_ih[j];
    for (int k = 0; k < E_; ++k)    a += x[k] * wi[k];
    for (int k = 0; k < TWOH_; ++k) a += c[k] * wi[E_ + k];
    gi[tid] = a;
    const float* h  = s_prev + (size_t)b * H_;
    const float* wh = w_hh + (size_t)j * H_;
    float ah = b_hh[j];
    for (int k = 0; k < H_; ++k) ah += h[k] * wh[k];
    gh[tid] = ah;
}

__global__ __launch_bounds__(256) void k_combine(const float* __restrict__ gi,
                                                 const float* __restrict__ gh,
                                                 const float* __restrict__ s_prev,
                                                 float* __restrict__ s_t,
                                                 half_t* __restrict__ sA) {
    int tid = blockIdx.x * 256 + threadIdx.x;       // < 32*256
    int b = tid >> 8, j = tid & 255;
    const float* gib = gi + (size_t)b * G3_;
    const float* ghb = gh + (size_t)b * G3_;
    float r = 1.f / (1.f + __expf(-(gib[j]        + ghb[j])));
    float z = 1.f / (1.f + __expf(-(gib[H_ + j]   + ghb[H_ + j])));
    float n = tanhf(gib[2 * H_ + j] + r * ghb[2 * H_ + j]);
    float s = (1.f - z) * n + z * s_prev[tid];
    s_t[tid] = s;
    sA[pack_idx(b >> 4, b & 15, j)] = (half_t)s;    // pack A operand for WMMA
}

// psi_g = s_t(32x256) @ Wg^T(256x32000): one wave per 16x16 output tile,
// 8 x v_wmma_f32_16x16x32_f16 over K=256.  4000 waves total (exact, EXEC all-1).
__global__ __launch_bounds__(256) void k_psi_g(const half_t* __restrict__ sA,
                                               const half_t* __restrict__ WgB,
                                               const float* __restrict__ Wg_b,
                                               float* __restrict__ logits) {
    int wave = blockIdx.x * (blockDim.x >> 5) + (threadIdx.x >> 5); // 0..3999
    int lane = threadIdx.x & 31;
    int ntile = wave >> 1;
    int mtile = wave & 1;
    const v16h* Ap = (const v16h*)sA;
    const v16h* Bp = (const v16h*)WgB;
    v8f c = {};
#pragma unroll
    for (int kc = 0; kc < 8; ++kc) {
        v16h a = Ap[(mtile * 8 + kc) * 32 + lane];
        v16h b = Bp[((size_t)ntile * 8 + kc) * 32 + lane];
        c = __builtin_amdgcn_wmma_f32_16x16x32_f16(false, a, false, b,
                                                   (short)0, c, false, false);
    }
    int col  = ntile * 16 + (lane & 15);
    float bias = Wg_b[col];
    int rbase = mtile * 16 + ((lane >> 4) << 3);
#pragma unroll
    for (int i = 0; i < 8; ++i)
        logits[(size_t)(rbase + i) * LG_ + col] = c[i] + bias;
}

__global__ __launch_bounds__(SEQ_) void k_psi_c(const float* __restrict__ es,
                                                const float* __restrict__ s_t,
                                                const int* __restrict__ eidx,
                                                float* __restrict__ logits) {
    int b = blockIdx.x, s = threadIdx.x;            // 32 x 128
    const float* e  = es  + ((size_t)b * SEQ_ + s) * H_;
    const float* st = s_t + (size_t)b * H_;
    float acc = 0.f;
    for (int k = 0; k < H_; ++k) acc += e[k] * st[k];
    if (eidx[b * SEQ_ + s] == 0) acc -= 1000.f;     // pad mask
    logits[(size_t)b * LG_ + V_ + s] = acc;
}

__global__ __launch_bounds__(256) void k_softmax(float* __restrict__ logits) {
    int b = blockIdx.x;
    float* row = logits + (size_t)b * LG_;
    __shared__ float red[256];
    float m = -1e30f;
    for (int i = threadIdx.x; i < LG_; i += 256) m = fmaxf(m, row[i]);
    red[threadIdx.x] = m; __syncthreads();
    for (int s = 128; s > 0; s >>= 1) {
        if (threadIdx.x < s) red[threadIdx.x] = fmaxf(red[threadIdx.x], red[threadIdx.x + s]);
        __syncthreads();
    }
    m = red[0]; __syncthreads();
    float sum = 0.f;
    for (int i = threadIdx.x; i < LG_; i += 256) sum += __expf(row[i] - m);
    red[threadIdx.x] = sum; __syncthreads();
    for (int s = 128; s > 0; s >>= 1) {
        if (threadIdx.x < s) red[threadIdx.x] += red[threadIdx.x + s];
        __syncthreads();
    }
    float inv = 1.f / red[0];
    for (int i = threadIdx.x; i < LG_; i += 256) row[i] = __expf(row[i] - m) * inv;
}

__global__ __launch_bounds__(256) void k_out_base(const float* __restrict__ p,
                                                  float* __restrict__ out, int t) {
    int tid = blockIdx.x * 256 + threadIdx.x;
    if (tid >= BS_ * VT_) return;
    int b = tid / VT_, v = tid % VT_;
    float val = (v < V_) ? p[(size_t)b * LG_ + v] : 1e-4f;
    out[((size_t)b * ML_ + t) * VT_ + v] = val;
}

// Deterministic copy-scatter: one thread per batch row, serial RMW (no atomics).
__global__ void k_scatter(const float* __restrict__ p, const int* __restrict__ eidx,
                          float* __restrict__ out, int t) {
    int b = threadIdx.x;                             // 32
    float* ob = out + ((size_t)b * ML_ + t) * VT_;
    const float* pb = p + (size_t)b * LG_ + V_;
    const int* ib = eidx + b * SEQ_;
    for (int s = 0; s < SEQ_; ++s) ob[ib[s]] += pb[s];
}

__global__ __launch_bounds__(SEQ_) void k_ctx(const float* __restrict__ p,
                                              const int* __restrict__ eidx,
                                              const int* __restrict__ inputs, int t,
                                              const float* __restrict__ encoded,
                                              float* __restrict__ ctx) {
    int b = blockIdx.x, s = threadIdx.x;            // 32 x 128
    __shared__ float cw[SEQ_];
    __shared__ float red[SEQ_];
    int tok = inputs[b * ML_ + t];
    float c = (eidx[b * SEQ_ + s] == tok) ? p[(size_t)b * LG_ + V_ + s] : 0.f;
    cw[s] = c; red[s] = c; __syncthreads();
    for (int st = 64; st > 0; st >>= 1) {
        if (s < st) red[s] += red[s + st];
        __syncthreads();
    }
    float inv = 1.f / fmaxf(red[0], 1e-12f);
    for (int d = s; d < TWOH_; d += SEQ_) {
        float acc = 0.f;
        const float* eb = encoded + (size_t)b * SEQ_ * TWOH_ + d;
        for (int q = 0; q < SEQ_; ++q) acc += cw[q] * eb[(size_t)q * TWOH_];
        ctx[(size_t)b * TWOH_ + d] = acc * inv;
    }
}

// ---------------------------------------------------------------------------

extern "C" void kernel_launch(void* const* d_in, const int* in_sizes, int n_in,
                              void* d_out, int out_size, void* d_ws, size_t ws_size,
                              hipStream_t stream) {
    (void)in_sizes; (void)n_in; (void)out_size; (void)ws_size;
    const int*   inputs   = (const int*)  d_in[0];
    const float* encoded  = (const float*)d_in[1];
    const int*   eidx     = (const int*)  d_in[2];
    const float* bridge_w = (const float*)d_in[3];
    const float* embed    = (const float*)d_in[4];
    const float* w_ih     = (const float*)d_in[5];
    const float* w_hh     = (const float*)d_in[6];
    const float* b_ih     = (const float*)d_in[7];
    const float* b_hh     = (const float*)d_in[8];
    const float* Wg_w     = (const float*)d_in[9];
    const float* Wg_b     = (const float*)d_in[10];
    const float* Wo_w     = (const float*)d_in[11];
    const float* Wo_b     = (const float*)d_in[12];
    float* out = (float*)d_out;

    char* ws = (char*)d_ws;
    size_t off = 0;
    auto alloc = [&](size_t bytes) -> void* {
        void* ptr = ws + off;
        off = (off + bytes + 255) & ~(size_t)255;
        return ptr;
    };
    half_t* WgB    = (half_t*)alloc((size_t)V_ * E_ * sizeof(half_t));   // 16.4 MB
    half_t* sA     = (half_t*)alloc((size_t)BS_ * H_ * sizeof(half_t));
    float*  logits = (float*)alloc((size_t)BS_ * LG_ * sizeof(float));   // 4.1 MB
    float*  escore = (float*)alloc((size_t)BS_ * SEQ_ * H_ * sizeof(float)); // 4.2 MB
    float*  gi     = (float*)alloc((size_t)BS_ * G3_ * sizeof(float));
    float*  gh     = (float*)alloc((size_t)BS_ * G3_ * sizeof(float));
    float*  sbuf0  = (float*)alloc((size_t)BS_ * H_ * sizeof(float));
    float*  sbuf1  = (float*)alloc((size_t)BS_ * H_ * sizeof(float));
    float*  ctx    = (float*)alloc((size_t)BS_ * TWOH_ * sizeof(float));

    // One-time: pack Wg_w -> f16 WMMA B-layout; enc_scores; s0 + ctx0.
    k_pack_wg<<<V_, 256, 0, stream>>>(Wg_w, WgB);           // 32000 blocks
    k_enc_scores<<<BS_ * SEQ_, 256, 0, stream>>>(encoded, Wo_w, Wo_b, escore);
    k_init<<<(BS_ * H_ + BS_ * TWOH_) / 256, 256, 0, stream>>>(encoded, bridge_w, sbuf0, ctx);

    for (int t = 0; t < ML_; ++t) {
        float* s_prev = (t & 1) ? sbuf1 : sbuf0;
        float* s_t    = (t & 1) ? sbuf0 : sbuf1;
        k_gru<<<(BS_ * G3_) / 256, 256, 0, stream>>>(inputs, t, embed, ctx, s_prev,
                                                     w_ih, w_hh, b_ih, b_hh, gi, gh);
        k_combine<<<(BS_ * H_) / 256, 256, 0, stream>>>(gi, gh, s_prev, s_t, sA);
        k_psi_g<<<500, 256, 0, stream>>>(sA, WgB, Wg_b, logits);    // 4000 waves
        k_psi_c<<<BS_, SEQ_, 0, stream>>>(escore, s_t, eidx, logits);
        k_softmax<<<BS_, 256, 0, stream>>>(logits);
        k_out_base<<<(BS_ * VT_ + 255) / 256, 256, 0, stream>>>(logits, out, t);
        k_scatter<<<1, BS_, 0, stream>>>(logits, eidx, out, t);
        k_ctx<<<BS_, SEQ_, 0, stream>>>(logits, eidx, inputs, t, encoded, ctx);
    }
}
